// MultiHeadedAttention_37228776521968
// MI455X (gfx1250) — matrix operand order repaired and checked by reading, not compiled, not run on hardware
//
#include <hip/hip_runtime.h>
#include <hip/hip_bf16.h>

#define B_  4
#define S_  1024
#define U_  1024
#define H_  16
#define DK_ 64

typedef __attribute__((ext_vector_type(16))) __bf16 v16bf;
typedef __attribute__((ext_vector_type(8)))  __bf16 v8bf;
typedef __attribute__((ext_vector_type(8)))  float  v8f;
typedef __attribute__((ext_vector_type(4)))  unsigned int u32x4;
typedef __attribute__((ext_vector_type(8)))  int i32x8;
typedef __attribute__((ext_vector_type(4)))  int i32x4;

__device__ __forceinline__ unsigned short f2bf(float f) {
    unsigned u = __float_as_uint(f);
    u += 0x7FFFu + ((u >> 16) & 1u);        // round-to-nearest-even
    return (unsigned short)(u >> 16);
}

// A-matrix fragment (16x32 bf16): per lane, two 8-half runs at +0 and +16 halves.
__device__ __forceinline__ v16bf fragA(const unsigned short* p) {
    v8bf lo = *(const v8bf*)(p);
    v8bf hi = *(const v8bf*)(p + 16);
    return __builtin_shufflevector(lo, hi, 0,1,2,3,4,5,6,7,8,9,10,11,12,13,14,15);
}
// B-matrix fragment (32x16 bf16): per lane, two 8-half runs at +0 and +16 halves.
__device__ __forceinline__ v16bf fragB(const unsigned short* p) {
    v8bf lo = *(const v8bf*)(p);
    v8bf hi = *(const v8bf*)(p + 16);
    return __builtin_shufflevector(lo, hi, 0,1,2,3,4,5,6,7,8,9,10,11,12,13,14,15);
}

#define WMMA_BF16(a, b, c) \
    __builtin_amdgcn_wmma_f32_16x16x32_bf16(false, (a), false, (b), (short)0, (c), false, false)

// ---- Tensor Data Mover: 2-D tile load (ISA 08_async_tensor §8).
// data_size code ds_code (3 = 8 bytes). tile_d0 elements per line, tile_d1 lines,
// stride0 = line stride (data_size units). tensor_dim == tile_dim -> no clipping.
// LDS receives the tile contiguously at lds_off. tile_d1==0 -> 1-D tile.
__device__ __forceinline__ void tdm_load(unsigned lds_off, const void* gptr,
                                         unsigned tile_d0, unsigned tile_d1,
                                         unsigned long long stride0,
                                         unsigned ds_code) {
    unsigned long long ga = (unsigned long long)(uintptr_t)gptr;
    u32x4 g0;
    g0[0] = 1u;                                                  // count=1
    g0[1] = lds_off;                                             // lds_addr
    g0[2] = (unsigned)ga;                                        // global_addr[31:0]
    g0[3] = (unsigned)((ga >> 32) & 0x01FFFFFFu) | (2u << 30);   // addr[56:32], type=2
    i32x8 g1;
    g1[0] = (int)(ds_code << 16);                                // mask=0, data_size
    g1[1] = (int)((tile_d0 & 0xFFFFu) << 16);                    // tensor_dim0[15:0]
    g1[2] = (int)(((tile_d0 >> 16) & 0xFFFFu) |                  // tensor_dim0[31:16]
                  ((tile_d1 & 0xFFFFu) << 16));                  // tensor_dim1[15:0]
    g1[3] = (int)(((tile_d1 >> 16) & 0xFFFFu) |                  // tensor_dim1[31:16]
                  ((tile_d0 & 0xFFFFu) << 16));                  // tile_dim0
    g1[4] = (int)(tile_d1 & 0xFFFFu);                            // tile_dim1, tile_dim2=0
    g1[5] = (int)(stride0 & 0xFFFFFFFFu);                        // dim0_stride[31:0]
    g1[6] = (int)((stride0 >> 32) & 0xFFFFu);                    // dim0_stride[47:32]
    g1[7] = 0;
    i32x4 z4 = {0, 0, 0, 0};
    i32x8 z8 = {0, 0, 0, 0, 0, 0, 0, 0};
    __builtin_amdgcn_tensor_load_to_lds(g0, g1, z4, z4, z8, 0);
}

// ---------------------------------------------------------------- prep kernels
__global__ __launch_bounds__(256) void conv_inputs_kernel(
        const float* __restrict__ q, const float* __restrict__ k,
        const float* __restrict__ v, unsigned short* __restrict__ X) {
    size_t i = (size_t)blockIdx.x * 256 + threadIdx.x;          // over 3*B*S*U
    const size_t n1 = (size_t)B_ * S_ * U_;
    const float* src; size_t off;
    if (i < n1)            { src = q; off = i; }
    else if (i < 2 * n1)   { src = k; off = i - n1; }
    else                   { src = v; off = i - 2 * n1; }
    X[i] = f2bf(src[off]);
}

// Wt[sel][n][k] = W[sel][h=n>>6][u=k][d=n&63]   (bf16, [N=1024][K=1024])
__global__ __launch_bounds__(256) void transpose_wqkv_kernel(
        const float* __restrict__ Wq, const float* __restrict__ Wk,
        const float* __restrict__ Wv, unsigned short* __restrict__ Wt) {
    size_t i = (size_t)blockIdx.x * 256 + threadIdx.x;          // over 3*1024*1024
    int sel = (int)(i >> 20);
    int r   = (int)(i & 1048575);
    int n = r >> 10, k = r & 1023;
    const float* W = sel == 0 ? Wq : (sel == 1 ? Wk : Wv);
    Wt[i] = f2bf(W[((size_t)(n >> 6) * 1024 + k) * 64 + (n & 63)]);
}

// WoT[n][k] = Wo[k][n]
__global__ __launch_bounds__(256) void transpose_wo_kernel(
        const float* __restrict__ Wo, unsigned short* __restrict__ WoT) {
    size_t i = (size_t)blockIdx.x * 256 + threadIdx.x;          // over 1024*1024
    int n = (int)(i >> 10), k = (int)(i & 1023);
    WoT[i] = f2bf(Wo[(size_t)k * 1024 + n]);
}

// ------------------------------------------------------- QKV projection GEMM
// C[m=(b,s)][n=(h,d)] = sum_u X[m][u] * Wt[n][u] (+bias) ; 128x64 block, BK=32
// TDM double-buffered tile staging: wave0 issues the DMA for stage i+1 while
// all 8 waves run WMMAs on stage i; TENSORcnt<=2 gates the release barrier.
__global__ __launch_bounds__(256) void gemm_qkv_kernel(
        const unsigned short* __restrict__ Xall,   // [3][B*S][U] bf16
        const unsigned short* __restrict__ Wt,     // [3][1024][1024] bf16
        const float* __restrict__ bq, const float* __restrict__ bk,
        const float* __restrict__ bv,
        unsigned short* __restrict__ qkv)          // [3][B][H][S][DK] bf16
{
    const int sel = blockIdx.z;
    const int m0 = blockIdx.y * 128;
    const int n0 = blockIdx.x * 64;
    const unsigned tid  = threadIdx.x;
    const unsigned lane = tid & 31u, wave = tid >> 5;
    const unsigned hi = (lane >> 4) & 1u, ln = lane & 15u;

    const unsigned short* A  = Xall + (size_t)sel * ((size_t)B_ * S_ * U_);
    const unsigned short* Bm = Wt   + (size_t)sel * (1024u * 1024u);
    const float* bias = sel == 0 ? bq : (sel == 1 ? bk : bv);

    __shared__ unsigned short lds_a[2][128 * 32];
    __shared__ unsigned short lds_b[2][64 * 32];

    v8f acc[2][2];
    #pragma unroll
    for (int i = 0; i < 2; ++i)
        #pragma unroll
        for (int j = 0; j < 2; ++j) acc[i][j] = (v8f){};

    const int wm = (wave & 3) * 32;
    const int wn = (wave >> 2) * 32;
    const int NIT = U_ / 32;

    if (tid < 32) {     // prologue: stage 0 into buffer 0
        tdm_load((unsigned)(uintptr_t)&lds_a[0][0], A + (size_t)m0 * U_, 8, 128, 256, 3);
        tdm_load((unsigned)(uintptr_t)&lds_b[0][0], Bm + (size_t)n0 * 1024, 8, 64, 256, 3);
    }

    for (int kk = 0; kk < NIT; ++kk) {
        const int cur = kk & 1;
        __syncthreads();                    // readers of buf[cur^1] (iter kk-1) done
        if (tid < 32) {
            if (kk + 1 < NIT) {
                const int k1 = (kk + 1) * 32;
                tdm_load((unsigned)(uintptr_t)&lds_a[cur ^ 1][0],
                         A + (size_t)m0 * U_ + k1, 8, 128, 256, 3);
                tdm_load((unsigned)(uintptr_t)&lds_b[cur ^ 1][0],
                         Bm + (size_t)n0 * 1024 + k1, 8, 64, 256, 3);
                __builtin_amdgcn_s_wait_tensorcnt((short)2);   // stage kk landed
            } else {
                __builtin_amdgcn_s_wait_tensorcnt((short)0);
            }
        }
        __syncthreads();
        #pragma unroll
        for (int tm = 0; tm < 2; ++tm) {
            v16bf af = fragA(&lds_a[cur][0] + (wm + tm * 16 + ln) * 32 + 8 * hi);
            #pragma unroll
            for (int tn = 0; tn < 2; ++tn) {
                v16bf bf = fragB(&lds_b[cur][0] + (wn + tn * 16 + ln) * 32 + 8 * hi);
                acc[tm][tn] = WMMA_BF16(af, bf, acc[tm][tn]);
            }
        }
    }

    #pragma unroll
    for (int tm = 0; tm < 2; ++tm)
        #pragma unroll
        for (int tn = 0; tn < 2; ++tn)
            #pragma unroll
            for (int r = 0; r < 8; ++r) {
                int m = m0 + wm + tm * 16 + r + 8 * (int)hi;
                int n = n0 + wn + tn * 16 + (int)ln;
                float val = acc[tm][tn][r] + bias[n];
                int b = m >> 10, s = m & 1023, h = n >> 6, d = n & 63;
                qkv[(((size_t)(sel * 4 + b) * 16 + h) * 1024 + s) * 64 + d] = f2bf(val);
            }
}

// ----------------------------------------------------- fused flash attention
// grid: (S/128, B*H); block 256 = 8 waves; wave w owns q-rows [w*16, w*16+16)
__global__ __launch_bounds__(256) void attn_kernel(
        const unsigned short* __restrict__ qkv,    // [3][B][H][S][DK] bf16
        const int* __restrict__ mask,              // [B][S][S]
        unsigned short* __restrict__ Hcat)         // [B][S][H*DK] bf16
{
    const int st = blockIdx.x;
    const int bh = blockIdx.y;
    const int b = bh >> 4, h = bh & 15;
    const unsigned tid  = threadIdx.x;
    const unsigned lane = tid & 31u, wave = tid >> 5;
    const unsigned hi = (lane >> 4) & 1u, ln = lane & 15u;

    const size_t hs = (size_t)S_ * DK_;
    const unsigned short* qg = qkv + ((size_t)0 * 64 + bh) * hs;
    const unsigned short* kg = qkv + ((size_t)1 * 64 + bh) * hs;
    const unsigned short* vg = qkv + ((size_t)2 * 64 + bh) * hs;

    // lds_kp: first 8192 halves = K tile [t][d] (TDM destination); later aliased
    // as 8 wave-private P stripes of 16x128 (barrier separates the uses).
    __shared__ unsigned short lds_kp[8 * 2048];    // 32 KB
    __shared__ unsigned short lds_vt[64 * 128];    // 16 KB, V transposed [d][t]

    // Q fragments held in registers for the whole kernel (A-matrix layout)
    v16bf qf[2];
    {
        const unsigned short* qrow = qg + (size_t)(st * 128 + wave * 16 + ln) * DK_;
        #pragma unroll
        for (int ks = 0; ks < 2; ++ks) {
            v8bf lo = *(const v8bf*)(qrow + ks * 32 + 8 * hi);
            v8bf hv = *(const v8bf*)(qrow + ks * 32 + 8 * hi + 16);
            qf[ks] = __builtin_shufflevector(lo, hv, 0,1,2,3,4,5,6,7,8,9,10,11,12,13,14,15);
        }
    }

    float mrow[8], lrow[8];
    v8f acc[4];
    #pragma unroll
    for (int r = 0; r < 8; ++r) { mrow[r] = -3.0e38f; lrow[r] = 0.0f; }
    #pragma unroll
    for (int dn = 0; dn < 4; ++dn) acc[dn] = (v8f){};

    const int srow_base = st * 128 + wave * 16;
    const float scale = 0.125f;                    // 1/sqrt(64)

    for (int t0 = 0; t0 < S_; t0 += 128) {
        __syncthreads();                           // prior P/Vt consumers done
        if (tid < 32)                              // K tile via TDM (contiguous 16 KB)
            tdm_load((unsigned)(uintptr_t)lds_kp, kg + (size_t)t0 * 64, 2048, 0, 2048, 3);
        for (int i = tid; i < 8192; i += 256) {    // V tile transposed: [d][t]
            int t = i >> 6, d = i & 63;            // (overlaps the TDM transfer)
            lds_vt[d * 128 + t] = vg[(size_t)(t0 + t) * 64 + d];
        }
        if (tid < 32) __builtin_amdgcn_s_wait_tensorcnt((short)0);
        __syncthreads();

        // ---- scores: 16x128 stripe per wave (8 n-tiles), K=64 in 2 steps
        v8f sc[8];
        #pragma unroll
        for (int tn = 0; tn < 8; ++tn) {
            v8f c = (v8f){};
            #pragma unroll
            for (int ks = 0; ks < 2; ++ks) {
                v16bf bf = fragB(lds_kp + (tn * 16 + ln) * 64 + ks * 32 + 8 * hi);
                c = WMMA_BF16(qf[ks], bf, c);
            }
            sc[tn] = c;
        }
        __syncthreads();                           // all waves done reading K tile

        // ---- scale + mask (reference: masked -> 1e-9 BEFORE softmax)
        float tmax[8];
        #pragma unroll
        for (int r = 0; r < 8; ++r) tmax[r] = -3.0e38f;
        #pragma unroll
        for (int tn = 0; tn < 8; ++tn) {
            int tcol = t0 + tn * 16 + (int)ln;
            #pragma unroll
            for (int r = 0; r < 8; ++r) {
                int srow = srow_base + r + 8 * (int)hi;
                float s = sc[tn][r] * scale;
                int mv = mask[((size_t)b * S_ + srow) * S_ + tcol];
                s = (mv == 0) ? 1e-9f : s;
                sc[tn][r] = s;
                tmax[r] = fmaxf(tmax[r], s);
            }
        }

        // ---- online softmax update (row reductions across 16-lane halves)
        #pragma unroll
        for (int r = 0; r < 8; ++r) {
            float tm = tmax[r];
            #pragma unroll
            for (int m = 1; m < 16; m <<= 1) tm = fmaxf(tm, __shfl_xor(tm, m, 32));
            float mnew = fmaxf(mrow[r], tm);
            float corr = __expf(mrow[r] - mnew);
            mrow[r] = mnew;
            float rsum = 0.0f;
            #pragma unroll
            for (int tn = 0; tn < 8; ++tn) {
                float p = __expf(sc[tn][r] - mnew);
                sc[tn][r] = p;
                rsum += p;
            }
            #pragma unroll
            for (int m = 1; m < 16; m <<= 1) rsum += __shfl_xor(rsum, m, 32);
            lrow[r] = lrow[r] * corr + rsum;
            #pragma unroll
            for (int dn = 0; dn < 4; ++dn) acc[dn][r] *= corr;
        }

        // ---- round-trip P through wave-private LDS (C-layout -> A-layout)
        unsigned short* pbuf = lds_kp + wave * 2048;
        #pragma unroll
        for (int tn = 0; tn < 8; ++tn)
            #pragma unroll
            for (int r = 0; r < 8; ++r)
                pbuf[(r + 8 * hi) * 128 + tn * 16 + ln] = f2bf(sc[tn][r]);

        // ---- acc += P (16x128) x V (128x64), K=128 in 4 steps
        #pragma unroll
        for (int kt = 0; kt < 4; ++kt) {
            v16bf af = fragA(pbuf + ln * 128 + kt * 32 + 8 * hi);
            #pragma unroll
            for (int dn = 0; dn < 4; ++dn) {
                v16bf bf = fragB(lds_vt + (dn * 16 + ln) * 128 + kt * 32 + 8 * hi);
                acc[dn] = WMMA_BF16(af, bf, acc[dn]);
            }
        }
    }

    // ---- normalize and store Hcat[b][s][h*64+d] (bf16 A-input of final GEMM)
    #pragma unroll
    for (int dn = 0; dn < 4; ++dn)
        #pragma unroll
        for (int r = 0; r < 8; ++r) {
            int srow = srow_base + r + 8 * (int)hi;
            int d = dn * 16 + (int)ln;
            float o = acc[dn][r] / lrow[r];
            Hcat[((size_t)b * S_ + srow) * U_ + h * 64 + d] = f2bf(o);
        }
}

// --------------------------------------------------------- output projection
__global__ __launch_bounds__(256) void gemm_out_kernel(
        const unsigned short* __restrict__ Hcat,   // [4096][1024] bf16
        const unsigned short* __restrict__ WoT,    // [1024][1024] bf16 [n][k]
        const float* __restrict__ bo,
        float* __restrict__ out)                   // [B][S][U] fp32
{
    const int m0 = blockIdx.y * 128;
    const int n0 = blockIdx.x * 64;
    const unsigned tid  = threadIdx.x;
    const unsigned lane = tid & 31u, wave = tid >> 5;
    const unsigned hi = (lane >> 4) & 1u, ln = lane & 15u;

    __shared__ unsigned short lds_a[2][128 * 32];
    __shared__ unsigned short lds_b[2][64 * 32];

    v8f acc[2][2];
    #pragma unroll
    for (int i = 0; i < 2; ++i)
        #pragma unroll
        for (int j = 0; j < 2; ++j) acc[i][j] = (v8f){};

    const int wm = (wave & 3) * 32;
    const int wn = (wave >> 2) * 32;
    const int NIT = U_ / 32;

    if (tid < 32) {
        tdm_load((unsigned)(uintptr_t)&lds_a[0][0], Hcat + (size_t)m0 * U_, 8, 128, 256, 3);
        tdm_load((unsigned)(uintptr_t)&lds_b[0][0], WoT + (size_t)n0 * 1024, 8, 64, 256, 3);
    }

    for (int kk = 0; kk < NIT; ++kk) {
        const int cur = kk & 1;
        __syncthreads();
        if (tid < 32) {
            if (kk + 1 < NIT) {
                const int k1 = (kk + 1) * 32;
                tdm_load((unsigned)(uintptr_t)&lds_a[cur ^ 1][0],
                         Hcat + (size_t)m0 * U_ + k1, 8, 128, 256, 3);
                tdm_load((unsigned)(uintptr_t)&lds_b[cur ^ 1][0],
                         WoT + (size_t)n0 * 1024 + k1, 8, 64, 256, 3);
                __builtin_amdgcn_s_wait_tensorcnt((short)2);
            } else {
                __builtin_amdgcn_s_wait_tensorcnt((short)0);
            }
        }
        __syncthreads();
        #pragma unroll
        for (int tm = 0; tm < 2; ++tm) {
            v16bf af = fragA(&lds_a[cur][0] + (wm + tm * 16 + ln) * 32 + 8 * hi);
            #pragma unroll
            for (int tn = 0; tn < 2; ++tn) {
                v16bf bf = fragB(&lds_b[cur][0] + (wn + tn * 16 + ln) * 32 + 8 * hi);
                acc[tm][tn] = WMMA_BF16(af, bf, acc[tm][tn]);
            }
        }
    }

    #pragma unroll
    for (int tm = 0; tm < 2; ++tm)
        #pragma unroll
        for (int tn = 0; tn < 2; ++tn)
            #pragma unroll
            for (int r = 0; r < 8; ++r) {
                int m = m0 + wm + tm * 16 + r + 8 * (int)hi;
                int n = n0 + wn + tn * 16 + (int)ln;
                out[(size_t)m * 1024 + n] = acc[tm][tn][r] + bo[n];
            }
}

// -----------------------------------------------------------------------------
extern "C" void kernel_launch(void* const* d_in, const int* in_sizes, int n_in,
                              void* d_out, int out_size, void* d_ws, size_t ws_size,
                              hipStream_t stream) {
    const float* query = (const float*)d_in[0];
    const float* key   = (const float*)d_in[1];
    const float* value = (const float*)d_in[2];
    const int*   mask  = (const int*)  d_in[3];
    const float* Wq    = (const float*)d_in[4];
    const float* bq    = (const float*)d_in[5];
    const float* Wk    = (const float*)d_in[6];
    const float* bk    = (const float*)d_in[7];
    const float* Wv    = (const float*)d_in[8];
    const float* bv    = (const float*)d_in[9];
    const float* Wo    = (const float*)d_in[10];
    const float* bo    = (const float*)d_in[11];
    float* out = (float*)d_out;

    char* ws = (char*)d_ws;
    unsigned short* X    = (unsigned short*)(ws + 0);          // 24 MB bf16 inputs
    unsigned short* Wt   = (unsigned short*)(ws + 25165824);   //  6 MB Wq/Wk/Wv^T
    unsigned short* WoT  = (unsigned short*)(ws + 31457280);   //  2 MB Wo^T
    unsigned short* QKV  = (unsigned short*)(ws + 33554432);   // 24 MB projected q/k/v
    unsigned short* Hcat = (unsigned short*)(ws + 58720256);   //  8 MB concat heads

    conv_inputs_kernel   <<<49152, 256, 0, stream>>>(query, key, value, X);
    transpose_wqkv_kernel<<<12288, 256, 0, stream>>>(Wq, Wk, Wv, Wt);
    transpose_wo_kernel  <<<4096,  256, 0, stream>>>(Wo, WoT);
    gemm_qkv_kernel      <<<dim3(16, 32, 3), 256, 0, stream>>>(X, Wt, bq, bk, bv, QKV);
    attn_kernel          <<<dim3(8, 64),     256, 0, stream>>>(QKV, mask, Hcat);
    gemm_out_kernel      <<<dim3(16, 32),    256, 0, stream>>>(Hcat, WoT, bo, out);
}
